// MagneticFieldGNN_47253230191459
// MI455X (gfx1250) — compile-verified, matrix operand order checked by reading, not run on hardware
//
#include <hip/hip_runtime.h>
#include <hip/hip_bf16.h>

#define NN 20000
#define NE 40000
#define BLKF 120   // per-node features: 32*1 + 16*3 + 8*5

typedef __attribute__((ext_vector_type(16))) __bf16 v16bf;
typedef __attribute__((ext_vector_type(8)))  float  v8f;

// ---------------- path tables ----------------
__constant__ int   c_pi1[11]  = {0,0,0,1,1,1,1,2,2,2,2};
__constant__ int   c_pi3[11]  = {0,1,2,1,0,2,1,2,1,0,2};
__constant__ int   c_pm1[11]  = {32,32,32,16,16,16,16,8,8,8,8};
__constant__ int   c_pm3[11]  = {32,16,8,16,32,8,16,8,16,32,8};
__constant__ int   c_poff[11] = {0,1024,1536,1792,2048,2560,2688,2944,3008,3136,3392};
__constant__ int   c_psho[11] = {0,1,4,0,1,1,4,0,1,4,4};
__constant__ float c_pscale[11] = {0.17677670f,0.30618622f,0.39528471f,0.43301270f,0.25f,
                                   0.55901699f,0.43301270f,0.79056942f,0.61237244f,
                                   0.35355339f,0.79056942f};
__constant__ int   c_xjoff[3] = {0,32,80};
__constant__ int   c_dim[3]   = {1,3,5};

// sparse real-Wigner-3j (normalized), 83 nonzeros over 11 paths
__constant__ int c_cgstart[12] = {0,1,4,9,12,15,26,37,42,53,58,83};
#define CG_R3 0.57735027f
#define CG_R5 0.44721360f
#define CG_A  0.31622777f
#define CG_B  0.18257419f
#define CG_B2 0.36514837f
#define CG_A2 0.23904572f
#define CG_B3 0.11952286f
#define CG_C3 0.20701967f
__constant__ signed char c_cgi[83] = {
  0, 0,0,0, 0,0,0,0,0, 0,1,2, 0,1,2,
  0,2,0,1,0,1,2,1,2,0,2,
  0,2,0,1,0,1,2,1,2,0,2,
  0,1,2,3,4,
  0,0,1,1,2,2,2,3,3,4,4,
  0,1,2,3,4,
  0,0,2, 1,1,2, 3,3,2, 4,4,2, 2, 1,1,4, 3,3,4, 0,0,1,1,3,3};
__constant__ signed char c_cgj[83] = {
  0, 0,1,2, 0,1,2,3,4, 0,0,0, 0,1,2,
  2,0,1,0,0,1,2,2,1,0,2,
  0,0,1,1,2,2,2,3,3,4,4,
  0,0,0,0,0,
  0,2,0,1,0,1,2,1,2,0,2,
  0,1,2,3,4,
  0,2,0, 1,2,1, 3,2,3, 4,2,4, 2, 1,4,1, 3,4,3, 1,3,0,3,0,1};
__constant__ signed char c_cgk[83] = {
  0, 0,1,2, 0,1,2,3,4, 0,1,2, 0,0,0,
  0,0,1,1,2,2,2,3,3,4,4,
  2,0,1,0,0,1,2,2,1,0,2,
  0,1,2,3,4,
  2,0,1,0,0,1,2,2,1,0,2,
  0,0,0,0,0,
  2,0,0, 2,1,1, 2,3,3, 2,4,4, 2, 4,1,1, 4,3,3, 3,1,3,0,1,0};
__constant__ float c_cgv[83] = {
  1.0f, CG_R3,CG_R3,CG_R3, CG_R5,CG_R5,CG_R5,CG_R5,CG_R5, CG_R3,CG_R3,CG_R3, CG_R3,CG_R3,CG_R3,
  CG_A,CG_A,CG_A,CG_A,-CG_B,CG_B2,-CG_B,CG_A,CG_A,-CG_A,CG_A,
  CG_A,CG_A,CG_A,CG_A,-CG_B,CG_B2,-CG_B,CG_A,CG_A,-CG_A,CG_A,
  CG_R5,CG_R5,CG_R5,CG_R5,CG_R5,
  CG_A,CG_A,CG_A,CG_A,-CG_B,CG_B2,-CG_B,CG_A,CG_A,-CG_A,CG_A,
  CG_R5,CG_R5,CG_R5,CG_R5,CG_R5,
  -CG_A2,-CG_A2,-CG_A2, CG_B3,CG_B3,CG_B3, CG_B3,CG_B3,CG_B3, -CG_A2,-CG_A2,-CG_A2, CG_A2,
  -CG_C3,-CG_C3,-CG_C3, CG_C3,CG_C3,CG_C3, CG_C3,CG_C3,CG_C3,CG_C3,CG_C3,CG_C3};
#undef CG_R3
#undef CG_R5
#undef CG_A
#undef CG_B
#undef CG_B2
#undef CG_A2
#undef CG_B3
#undef CG_C3

// ---------------- utility kernels ----------------
__global__ void zero_kernel(float* p, int n) {
  int i = blockIdx.x * blockDim.x + threadIdx.x;
  if (i < n) p[i] = 0.f;
}

__global__ void sph_kernel(const float* __restrict__ ea, float* __restrict__ sh) {
  int e = blockIdx.x * blockDim.x + threadIdx.x;
  if (e >= NE) return;
  float vx = ea[e*4+0], vy = ea[e*4+1], vz = ea[e*4+2];
  float rn = 1.f / (sqrtf(vx*vx + vy*vy + vz*vz) + 1e-12f);
  float x = vx*rn, y = vy*rn, z = vz*rn;
  const float s3 = 1.7320508f, s5 = 2.2360680f, s15 = 3.8729833f;
  float* o = sh + e*9;
  o[0] = 1.f;      o[1] = s3*y;     o[2] = s3*z;     o[3] = s3*x;
  o[4] = s15*x*y;  o[5] = s15*y*z;  o[6] = 0.5f*s5*(3.f*z*z - 1.f);
  o[7] = s15*x*z;  o[8] = 0.5f*s15*(x*x - y*y);
}

__global__ void embed_kernel(const float* __restrict__ x, const float* __restrict__ w0,
                             const float* __restrict__ w1, float* __restrict__ blk) {
  int tid = blockIdx.x * blockDim.x + threadIdx.x;
  if (tid >= NN*BLKF) return;
  int n = tid / BLKF, c = tid % BLKF;
  float s;
  if (c < 32) {
    s = (x[n*5+0]*w0[c] + x[n*5+1]*w0[32+c]) * 0.70710678f;
  } else if (c < 80) {
    int r = c - 32, v = r/3, i = r%3;
    s = x[n*5+2+i] * w1[v];
  } else s = 0.f;
  blk[tid] = s;
}

// pack fp32 KxNcols weights into bf16 WMMA B-fragment order:
// idx = ((ct*nks + s)*32 + lane)*16 + elem
__global__ void pack_kernel(const float* __restrict__ W, __bf16* __restrict__ out,
                            int ncols, int nks) {
  int idx = blockIdx.x * blockDim.x + threadIdx.x;
  int total = (ncols/16) * nks * 512;
  if (idx >= total) return;
  int elem = idx & 15;
  int lane = (idx >> 4) & 31;
  int s    = (idx >> 9) % nks;
  int ct   = idx / (512 * nks);
  int n  = ct*16 + (lane & 15);
  int kb = (lane < 16) ? 0 : 8;
  int k  = 32*s + ((elem < 8) ? elem : elem + 8) + kb;
  out[idx] = (__bf16)W[k*ncols + n];
}

// mixed = (agg + hid_linear(in, selfW))/sqrt(20)   (addAgg=1)
// out   = hid_linear(in, linW)                     (addAgg=0)
__global__ void hidlin_kernel(const float* __restrict__ in, const float* __restrict__ agg,
                              const float* __restrict__ W0, const float* __restrict__ W1m,
                              const float* __restrict__ W2m, float* __restrict__ out,
                              int addAgg) {
  int tid = blockIdx.x * blockDim.x + threadIdx.x;
  if (tid >= NN*BLKF) return;
  int n = tid / BLKF, c = tid % BLKF;
  const float* inr = in + n*BLKF;
  float s = 0.f;
  if (c < 32) {
    for (int u = 0; u < 32; ++u) s += inr[u] * W0[u*32 + c];
    s *= 0.17677670f;
  } else if (c < 80) {
    int r = c - 32, v = r/3, i = r%3;
    for (int u = 0; u < 16; ++u) s += inr[32 + u*3 + i] * W1m[u*16 + v];
    s *= 0.25f;
  } else {
    int r = c - 80, v = r/5, k = r%5;
    for (int u = 0; u < 8; ++u) s += inr[80 + u*5 + k] * W2m[u*8 + v];
    s *= 0.35355339f;
  }
  if (addAgg) s = (s + agg[tid]) * 0.22360680f;  // 1/sqrt(20)
  out[tid] = s;
}

__global__ void out_kernel(const float* __restrict__ blk, const float* __restrict__ outW,
                           float* __restrict__ out) {
  int tid = blockIdx.x * blockDim.x + threadIdx.x;
  if (tid >= NN*3) return;
  int n = tid/3, i = tid%3;
  float s = 0.f;
  for (int u = 0; u < 16; ++u) s += blk[n*BLKF + 32 + u*3 + i] * outW[u];
  out[tid] = s * 0.25f;
}

// ---------------- fused edge kernel: fc_net (WMMA) + tensor product + scatter ----------------
__global__ __launch_bounds__(32)
void edge_msg_kernel(const float* __restrict__ sh_g, const int* __restrict__ edge_index,
                     const float* __restrict__ blocks, float* __restrict__ agg,
                     const float* __restrict__ W1, const float* __restrict__ b1,
                     const __bf16* __restrict__ pW2, const float* __restrict__ b2,
                     const __bf16* __restrict__ pW3, const float* __restrict__ b3) {
  __shared__ float shb[16][9];
  __shared__ float xjb[16][BLKF];
  __shared__ float h1[16][64];
  __shared__ float h2[16][64];
  __shared__ float wtile[16][16];
  __shared__ float tbuf[16*32*5];   // max 16*m1*d3
  __shared__ float obuf[16*48];     // max 16*m3*d3
  __shared__ float W1s[9*64];
  __shared__ float b1s[64];
  __shared__ float b2s[64];

  const int lane = threadIdx.x;
  const int e0 = blockIdx.x * 16;
  const int* srcI = edge_index;
  const int* dstI = edge_index + NE;

  for (int i = lane; i < 16*9;     i += 32) shb[i/9][i%9] = sh_g[(e0 + i/9)*9 + (i%9)];
  for (int i = lane; i < 16*BLKF;  i += 32) {
    int e = i / BLKF, c = i % BLKF;
    xjb[e][c] = blocks[(long)srcI[e0+e]*BLKF + c];
  }
  for (int i = lane; i < 9*64; i += 32) W1s[i] = W1[i];
  for (int i = lane; i < 64;   i += 32) { b1s[i] = b1[i]; b2s[i] = b2[i]; }
  __syncthreads();

  // --- h1 = silu(sh @ W1 + b1)  (K=9: plain FMA) ---
  {
    int e = lane & 15, half = lane >> 4;
    for (int c = half*32; c < half*32 + 32; ++c) {
      float a = b1s[c];
      for (int i = 0; i < 9; ++i) a += shb[e][i] * W1s[i*64 + c];
      h1[e][c] = a / (1.f + __expf(-a));
    }
  }
  __syncthreads();

  // --- h2 = silu(h1 @ W2 + b2)  via WMMA bf16 ---
  {
    v16bf a0, a1;
    int e = lane & 15, base = (lane < 16) ? 0 : 8;
    for (int i = 0; i < 16; ++i) {
      int k = ((i < 8) ? i : i + 8) + base;
      a0[i] = (__bf16)h1[e][k];
      a1[i] = (__bf16)h1[e][32 + k];
    }
    for (int ct = 0; ct < 4; ++ct) {
      v8f acc = {};
      v16bf bA = *(const v16bf*)(pW2 + ((ct*2 + 0)*32 + lane)*16);
      v16bf bB = *(const v16bf*)(pW2 + ((ct*2 + 1)*32 + lane)*16);
      acc = __builtin_amdgcn_wmma_f32_16x16x32_bf16(false, a0, false, bA, (short)0, acc, false, false);
      acc = __builtin_amdgcn_wmma_f32_16x16x32_bf16(false, a1, false, bB, (short)0, acc, false, false);
      int col = ct*16 + (lane & 15);
      int rb  = (lane < 16) ? 0 : 8;
      for (int r = 0; r < 8; ++r) {
        float v = acc[r] + b2s[col];
        h2[rb + r][col] = v / (1.f + __expf(-v));
      }
    }
  }
  __syncthreads();

  // --- A fragments of h2 (shared by all paths / column tiles) ---
  v16bf Afrag0, Afrag1;
  {
    int e = lane & 15, base = (lane < 16) ? 0 : 8;
    for (int i = 0; i < 16; ++i) {
      int k = ((i < 8) ? i : i + 8) + base;
      Afrag0[i] = (__bf16)h2[e][k];
      Afrag1[i] = (__bf16)h2[e][32 + k];
    }
  }

  // --- per-path: WMMA weight tiles fused with tensor-product contraction ---
  for (int p = 0; p < 11; ++p) {
    const int i1 = c_pi1[p], i3 = c_pi3[p];
    const int m1 = c_pm1[p], m3 = c_pm3[p];
    const int d1 = c_dim[i1], d3 = c_dim[i3];
    const int off = c_poff[p], sho = c_psho[p];

    for (int i = lane; i < 16*m1*d3; i += 32) tbuf[i] = 0.f;
    for (int i = lane; i < 16*m3*d3; i += 32) obuf[i] = 0.f;
    __syncthreads();

    // t[e,u,k] = sum_{ij} xj[e,u,i] * sh[e,j] * CG[i,j,k]  (sparse CG)
    for (int idx = lane; idx < 16*m1; idx += 32) {
      int e = idx / m1, u = idx % m1;
      int tb = (e*m1 + u) * d3;
      const float* xb  = &xjb[e][c_xjoff[i1] + u*d1];
      const float* shp = &shb[e][sho];
      for (int s = c_cgstart[p]; s < c_cgstart[p+1]; ++s)
        tbuf[tb + c_cgk[s]] += xb[(int)c_cgi[s]] * shp[(int)c_cgj[s]] * c_cgv[s];
    }
    __syncthreads();

    const int ntiles = (m1*m3) >> 4;
    const int gt0 = off >> 4;
    for (int ct = 0; ct < ntiles; ++ct) {
      v8f acc = {};
      v16bf bA = *(const v16bf*)(pW3 + (((gt0 + ct)*2 + 0)*32 + lane)*16);
      v16bf bB = *(const v16bf*)(pW3 + (((gt0 + ct)*2 + 1)*32 + lane)*16);
      acc = __builtin_amdgcn_wmma_f32_16x16x32_bf16(false, Afrag0, false, bA, (short)0, acc, false, false);
      acc = __builtin_amdgcn_wmma_f32_16x16x32_bf16(false, Afrag1, false, bB, (short)0, acc, false, false);
      int col = lane & 15;
      int rb  = (lane < 16) ? 0 : 8;
      float bb = b3[off + ct*16 + col];
      __syncthreads();
      for (int r = 0; r < 8; ++r) wtile[rb + r][col] = acc[r] + bb;
      __syncthreads();
      // obuf[e,v,k] += t[e,u,k] * w[e, u*m3+v]
      for (int cc = 0; cc < 16; ++cc) {
        int g = ct*16 + cc, u = g / m3, v = g % m3;
        for (int idx = lane; idx < 16*d3; idx += 32) {
          int e = idx / d3, k = idx % d3;
          obuf[(e*m3 + v)*d3 + k] += tbuf[(e*m1 + u)*d3 + k] * wtile[e][cc];
        }
      }
    }
    __syncthreads();

    // scatter messages into node aggregates
    const float sc = c_pscale[p];
    const int aoff = c_xjoff[i3];
    for (int idx = lane; idx < 16*m3*d3; idx += 32) {
      int e = idx / (m3*d3), r = idx % (m3*d3);
      atomicAdd(&agg[(long)dstI[e0 + e]*BLKF + aoff + r], sc * obuf[idx]);
    }
    __syncthreads();
  }
}

// ---------------- host ----------------
extern "C" void kernel_launch(void* const* d_in, const int* in_sizes, int n_in,
                              void* d_out, int out_size, void* d_ws, size_t ws_size,
                              hipStream_t stream) {
  (void)in_sizes; (void)n_in; (void)out_size; (void)ws_size;
  const float* x     = (const float*)d_in[0];
  const int*   ei    = (const int*)  d_in[1];
  const float* ea    = (const float*)d_in[2];
  const float* embW0 = (const float*)d_in[3];
  const float* embW1 = (const float*)d_in[4];
  struct Layer { const float *sW0,*sW1,*sW2,*lW0,*lW1,*lW2,*W1,*b1,*W2,*b2,*W3,*b3; };
  Layer L[2];
  for (int l = 0; l < 2; ++l) {
    int b = 5 + l*12;
    L[l].sW0 = (const float*)d_in[b+0];  L[l].sW1 = (const float*)d_in[b+1];
    L[l].sW2 = (const float*)d_in[b+2];  L[l].lW0 = (const float*)d_in[b+3];
    L[l].lW1 = (const float*)d_in[b+4];  L[l].lW2 = (const float*)d_in[b+5];
    L[l].W1  = (const float*)d_in[b+6];  L[l].b1  = (const float*)d_in[b+7];
    L[l].W2  = (const float*)d_in[b+8];  L[l].b2  = (const float*)d_in[b+9];
    L[l].W3  = (const float*)d_in[b+10]; L[l].b3  = (const float*)d_in[b+11];
  }
  const float* outW = (const float*)d_in[29];

  float* ws    = (float*)d_ws;
  float* sh    = ws;                       // 360000
  float* blkA  = sh    + 360000;           // NN*120
  float* blkB  = blkA  + NN*BLKF;
  float* mixed = blkB  + NN*BLKF;
  float* agg   = mixed + NN*BLKF;
  __bf16* packs = (__bf16*)(agg + NN*BLKF);
  __bf16* pW2[2] = {packs,        packs + 4096};
  __bf16* pW3[2] = {packs + 8192, packs + 8192 + 221184};

  sph_kernel  <<<(NE + 255)/256, 256, 0, stream>>>(ea, sh);
  embed_kernel<<<(NN*BLKF + 255)/256, 256, 0, stream>>>(x, embW0, embW1, blkA);
  for (int l = 0; l < 2; ++l) {
    pack_kernel<<<(4*2*512   + 255)/256, 256, 0, stream>>>(L[l].W2, pW2[l], 64,   2);
    pack_kernel<<<(216*2*512 + 255)/256, 256, 0, stream>>>(L[l].W3, pW3[l], 3456, 2);
  }

  float* cur = blkA; float* nxt = blkB;
  for (int l = 0; l < 2; ++l) {
    zero_kernel<<<(NN*BLKF + 255)/256, 256, 0, stream>>>(agg, NN*BLKF);
    edge_msg_kernel<<<NE/16, 32, 0, stream>>>(sh, ei, cur, agg,
        L[l].W1, L[l].b1, pW2[l], L[l].b2, pW3[l], L[l].b3);
    hidlin_kernel<<<(NN*BLKF + 255)/256, 256, 0, stream>>>(cur, agg,
        L[l].sW0, L[l].sW1, L[l].sW2, mixed, 1);
    hidlin_kernel<<<(NN*BLKF + 255)/256, 256, 0, stream>>>(mixed, agg,
        L[l].lW0, L[l].lW1, L[l].lW2, nxt, 0);
    float* t = cur; cur = nxt; nxt = t;
  }
  out_kernel<<<(NN*3 + 255)/256, 256, 0, stream>>>(cur, outW, (float*)d_out);
}
